// ISTA2_38302518346314
// MI455X (gfx1250) — matrix operand order repaired and checked by reading, not compiled, not run on hardware
//
#include <hip/hip_runtime.h>
#include <hip/hip_bf16.h>

// ---------------- problem constants ----------------
#define B_   32
#define P_   197
#define D_   768
#define H_   12
#define d_   64
#define Pp_  49
#define NREF 16          // B * BS_RATIO
#define T_   784         // NREF * Pp
#define PK_  224         // j padded to multiple of 32
#define NT_  13          // ceil(197/16) row tiles
#define TOPK_ 8
#define NEGV (-1e30f)

typedef __bf16 bf16_t;
typedef __attribute__((ext_vector_type(16))) __bf16 v16bf;
typedef __attribute__((ext_vector_type(8)))  __bf16 v8bf;
typedef __attribute__((ext_vector_type(8)))  float  v8f;

// ---- WMMA helpers (CDNA5 gfx1250, wave32) ----
__device__ __forceinline__ v8f wmma_bf16(v16bf a, v16bf b, v8f c) {
  return __builtin_amdgcn_wmma_f32_16x16x32_bf16(false, a, false, b, (short)0, c, false, false);
}
// A 16x32 fragment: lane holds row M=lane&15; half=lane>>4 picks K groups.
// Elements e=0..7 -> K = half*8+e ; e=8..15 -> K = 16+half*8+(e-8)  (ISA 7.12.2)
__device__ __forceinline__ v16bf load_a_frag(const bf16_t* rowK0, int half) {
  v16bf f;
  ((v8bf*)&f)[0] = *(const v8bf*)(rowK0 + half * 8);
  ((v8bf*)&f)[1] = *(const v8bf*)(rowK0 + 16 + half * 8);
  return f;
}
// B 32x16 fragment: lane holds col N=lane&15; K = half*8+e (e<8), 16+half*8+(e-8) (e>=8)
__device__ __forceinline__ v16bf load_b_frag(const bf16_t* colK0, int half) {
  v16bf f;
  ((v8bf*)&f)[0] = *(const v8bf*)(colK0 + half * 8);
  ((v8bf*)&f)[1] = *(const v8bf*)(colK0 + 16 + half * 8);
  return f;
}
// wave32 butterfly reductions (all lanes receive result)
__device__ __forceinline__ float wave_max(float v) {
#pragma unroll
  for (int off = 16; off; off >>= 1) v = fmaxf(v, __shfl_xor(v, off, 32));
  return v;
}
__device__ __forceinline__ float wave_sum(float v) {
#pragma unroll
  for (int off = 16; off; off >>= 1) v += __shfl_xor(v, off, 32);
  return v;
}

// ---------------- kernel 1: RMSNorm + head split + bf16 cast ----------------
__global__ void k_rmsnorm(const float* __restrict__ q, const float* __restrict__ k,
                          const float* __restrict__ v, const float* __restrict__ wq,
                          const float* __restrict__ wk, const float* __restrict__ wv,
                          bf16_t* __restrict__ qh, bf16_t* __restrict__ kh,
                          bf16_t* __restrict__ vh, float* __restrict__ vhf) {
  const int bp = blockIdx.x;            // b*P + p
  const int b  = bp / P_;
  const int p  = bp % P_;
  const int tid = threadIdx.x;
  const float* qr = q + (size_t)bp * D_;
  const float* kr = k + (size_t)bp * D_;
  const float* vr = v + (size_t)bp * D_;

  float lq[3], lk[3], lv[3];
  float sq = 0.f, sk = 0.f, sv = 0.f;
#pragma unroll
  for (int i = 0; i < 3; ++i) {
    int dd = tid + i * 256;
    lq[i] = qr[dd]; lk[i] = kr[dd]; lv[i] = vr[dd];
    sq += lq[i] * lq[i]; sk += lk[i] * lk[i]; sv += lv[i] * lv[i];
  }
  __shared__ float red[256];
  float sums[3] = {sq, sk, sv};
#pragma unroll
  for (int a = 0; a < 3; ++a) {
    float w = wave_sum(sums[a]);                 // intra-wave first
    if ((tid & 31) == 0) red[tid >> 5] = w;
    __syncthreads();
    if (tid < 8) {
      float t = red[tid];
#pragma unroll
      for (int off = 4; off; off >>= 1) t += __shfl_xor(t, off, 32);
      if (tid == 0) red[0] = t;
    }
    __syncthreads();
    sums[a] = red[0];
    __syncthreads();
  }
  const float rq = __frsqrt_rn(sums[0] * (1.0f / D_) + 1e-6f);
  const float rk = __frsqrt_rn(sums[1] * (1.0f / D_) + 1e-6f);
  const float rv = __frsqrt_rn(sums[2] * (1.0f / D_) + 1e-6f);

#pragma unroll
  for (int i = 0; i < 3; ++i) {
    int dd = tid + i * 256;
    int h  = dd >> 6;
    int dl = dd & 63;
    size_t o = (((size_t)(b * H_ + h) * P_ + p) << 6) + dl;
    float vq = lq[i] * rq * wq[dd];
    float vk = lk[i] * rk * wk[dd];
    float vv = lv[i] * rv * wv[dd];
    qh[o] = (bf16_t)vq;
    kh[o] = (bf16_t)vk;
    vh[o] = (bf16_t)vv;
    vhf[o] = vv;
  }
}

// ---------------- kernel 2: representative-token gather ----------------
__global__ void k_repgather(const float* __restrict__ vhf, const int* __restrict__ idxs,
                            bf16_t* __restrict__ repb, float* __restrict__ repf) {
  int i = blockIdx.x * 256 + threadIdx.x;          // over H*T*64
  if (i >= H_ * T_ * 64) return;
  int dd = i & 63;
  int t  = (i >> 6) % T_;
  int h  = (i >> 6) / T_;
  int nn = t / Pp_;
  int jj = t - nn * Pp_;
  int pi = idxs[jj];
  if (pi < 0) pi = 0; if (pi >= P_) pi = P_ - 1;
  float val = vhf[(((size_t)(nn * H_ + h) * P_ + pi) << 6) + dd];
  repf[i] = val;
  repb[i] = (bf16_t)val;
}

// ---------------- kernel 2b: zero v_combT (padding determinism) ----------------
__global__ void k_zero16(unsigned short* __restrict__ p, int n) {
  int i = blockIdx.x * 256 + threadIdx.x;
  if (i < n) p[i] = 0;
}

// ---------------- kernel 3: QK^T + softmax -> attn (bf16, j padded to 224) ----
__global__ void k_attn(const bf16_t* __restrict__ qh, const bf16_t* __restrict__ kh,
                       bf16_t* __restrict__ attn) {
  const int gid = blockIdx.x;
  const int it  = gid % NT_;
  const int bh  = gid / NT_;
  __shared__ float S[16][PK_];
  __shared__ float rmax[16], rinv[16];
  const int tid  = threadIdx.x;
  const int lane = tid & 31;
  const int wv   = tid >> 5;       // 8 waves
  const int half = lane >> 4;
  const int l15  = lane & 15;

  const bf16_t* qbase = qh + (size_t)bh * P_ * 64;
  const bf16_t* kbase = kh + (size_t)bh * P_ * 64;

  int arow = it * 16 + l15; if (arow >= P_) arow = P_ - 1;
  v16bf a0 = load_a_frag(qbase + (size_t)arow * 64, half);
  v16bf a1 = load_a_frag(qbase + (size_t)arow * 64 + 32, half);

  for (int jt = wv; jt < NT_; jt += 8) {
    int bcol = jt * 16 + l15; if (bcol >= P_) bcol = P_ - 1;
    v16bf b0 = load_b_frag(kbase + (size_t)bcol * 64, half);
    v16bf b1 = load_b_frag(kbase + (size_t)bcol * 64 + 32, half);
    v8f acc = {};
    acc = wmma_bf16(a0, b0, acc);
    acc = wmma_bf16(a1, b1, acc);
#pragma unroll
    for (int r = 0; r < 8; ++r) {
      int row = r + half * 8;
      S[row][jt * 16 + l15] = acc[r] * 0.125f;   // QK_SCALE * d^-0.5
    }
  }
  __syncthreads();
  // one wave per row (rows wv, wv+8): butterfly max/sum over strided cols
  for (int r = wv; r < 16; r += 8) {
    float m = -3e38f;
    for (int j = lane; j < P_; j += 32) m = fmaxf(m, S[r][j]);
    m = wave_max(m);
    float s = 0.f;
    for (int j = lane; j < P_; j += 32) s += __expf(S[r][j] - m);
    s = wave_sum(s);
    if (lane == 0) { rmax[r] = m; rinv[r] = 1.0f / s; }
  }
  __syncthreads();
  bf16_t* ab = attn + (size_t)bh * P_ * PK_;
  for (int idx = tid; idx < 16 * PK_; idx += 256) {
    int r = idx / PK_, j = idx % PK_;
    int gi = it * 16 + r;
    if (gi >= P_) continue;
    float pv = (j < P_) ? __expf(S[r][j] - rmax[r]) * rinv[r] : 0.0f;
    ab[(size_t)gi * PK_ + j] = (bf16_t)pv;
  }
}

// ---- kernel 4: sim GEMM + self-mask + top-8 sparse softmax + v_aligned + combine
__global__ void k_sim(const bf16_t* __restrict__ vhb, const bf16_t* __restrict__ repb,
                      const float* __restrict__ repf, const float* __restrict__ vhf,
                      bf16_t* __restrict__ vcombT) {
  const int gid = blockIdx.x;
  const int it  = gid % NT_;
  const int bh  = gid / NT_;
  const int b   = bh / H_;
  const int h   = bh % H_;
  __shared__ float S[16][T_];                  // 16*784*4 = 50176 B
  __shared__ int   kcnt[16];
  __shared__ int   kidx[16][64];
  __shared__ float kwt[16][64];

  const int tid  = threadIdx.x;
  const int lane = tid & 31;
  const int wv   = tid >> 5;
  const int half = lane >> 4;
  const int l15  = lane & 15;

  if (tid < 16) kcnt[tid] = 0;

  const bf16_t* vbase = vhb + (size_t)bh * P_ * 64;
  const bf16_t* rbase = repb + (size_t)h * T_ * 64;

  int arow = it * 16 + l15; if (arow >= P_) arow = P_ - 1;
  v16bf a0 = load_a_frag(vbase + (size_t)arow * 64, half);
  v16bf a1 = load_a_frag(vbase + (size_t)arow * 64 + 32, half);

  for (int tt = wv; tt < T_ / 16; tt += 8) {          // 49 column tiles
    int col = tt * 16 + l15;
    v16bf b0 = load_b_frag(rbase + (size_t)col * 64, half);
    v16bf b1 = load_b_frag(rbase + (size_t)col * 64 + 32, half);
    v8f acc = {};
    acc = wmma_bf16(a0, b0, acc);
    acc = wmma_bf16(a1, b1, acc);
    const bool masked = (b < NREF) && (col / Pp_ == b); // exclude own tokens
#pragma unroll
    for (int r = 0; r < 8; ++r) {
      int row = r + half * 8;
      S[row][col] = masked ? NEGV : acc[r];            // V_SCALE == 1
    }
  }
  __syncthreads();

  // one wave per row: lane-local top-8 over strided scan, then 8x pop-max merge
  for (int r = wv; r < 16; r += 8) {
    float loc[TOPK_];
#pragma unroll
    for (int i = 0; i < TOPK_; ++i) loc[i] = -3e38f;
    for (int t = lane; t < T_; t += 32) {
      float v = S[r][t];
      if (v > loc[TOPK_ - 1]) {
        int pos = TOPK_ - 1;
        while (pos > 0 && v > loc[pos - 1]) { loc[pos] = loc[pos - 1]; --pos; }
        loc[pos] = v;
      }
    }
    int li = 0;
    float mx = -3e38f, thr = -3e38f;
#pragma unroll
    for (int k2 = 0; k2 < TOPK_; ++k2) {
      float head = (li < TOPK_) ? loc[li] : -3e38f;
      float g = wave_max(head);
      if (li < TOPK_ && loc[li] == g) ++li;
      if (k2 == 0) mx = g;
      thr = g;
    }
    float s = 0.f;
    for (int t = lane; t < T_; t += 32) {
      float v = S[r][t];
      if (v >= thr) s += __expf(v - mx);
    }
    s = wave_sum(s);
    const float inv = 1.0f / s;
    for (int t = lane; t < T_; t += 32) {
      float v = S[r][t];
      if (v >= thr) {
        int slot = atomicAdd(&kcnt[r], 1);
        if (slot < 64) { kidx[r][slot] = t; kwt[r][slot] = __expf(v - mx) * inv; }
      }
    }
  }
  __syncthreads();

  // v_comb = vh + v_aligned ; write transposed (d, j_padded) bf16 for final GEMM
  const int row = tid >> 4;
  const int dg  = tid & 15;
  const int p   = it * 16 + row;
  if (p < P_) {
    float acc4[4];
    const float* vsrc = vhf + (((size_t)bh * P_ + p) << 6) + dg * 4;
#pragma unroll
    for (int c = 0; c < 4; ++c) acc4[c] = vsrc[c];
    int cnt = kcnt[row]; if (cnt > 64) cnt = 64;
    const float* rf = repf + (size_t)h * T_ * 64 + dg * 4;
    for (int i = 0; i < cnt; ++i) {
      const float w = kwt[row][i];
      const float* rr = rf + (size_t)kidx[row][i] * 64;
#pragma unroll
      for (int c = 0; c < 4; ++c) acc4[c] += w * rr[c];
    }
    bf16_t* dst = vcombT + (size_t)bh * 64 * PK_;
#pragma unroll
    for (int c = 0; c < 4; ++c)
      dst[(size_t)(dg * 4 + c) * PK_ + p] = (bf16_t)acc4[c];
  }
}

// ---------------- kernel 5: out = attn @ v_comb ----------------
__global__ void k_out(const bf16_t* __restrict__ attn, const bf16_t* __restrict__ vcombT,
                      float* __restrict__ out) {
  const int bh = blockIdx.x;
  const int b  = bh / H_;
  const int h  = bh % H_;
  const int tid  = threadIdx.x;
  const int lane = tid & 31;
  const int wv   = tid >> 5;
  const int half = lane >> 4;
  const int l15  = lane & 15;

  const bf16_t* abase = attn + (size_t)bh * P_ * PK_;
  const bf16_t* vb    = vcombT + (size_t)bh * 64 * PK_;

  for (int it = wv; it < NT_; it += 8) {
    int arow = it * 16 + l15; if (arow >= P_) arow = P_ - 1;
    v16bf af[PK_ / 32];
#pragma unroll
    for (int kc = 0; kc < PK_ / 32; ++kc)
      af[kc] = load_a_frag(abase + (size_t)arow * PK_ + kc * 32, half);
#pragma unroll
    for (int nt = 0; nt < 4; ++nt) {
      const int dd = nt * 16 + l15;
      v8f acc = {};
#pragma unroll
      for (int kc = 0; kc < PK_ / 32; ++kc) {
        v16bf bf = load_b_frag(vb + (size_t)dd * PK_ + kc * 32, half);
        acc = wmma_bf16(af[kc], bf, acc);
      }
#pragma unroll
      for (int r = 0; r < 8; ++r) {
        int p = it * 16 + r + half * 8;
        if (p < P_)
          out[((size_t)(b * P_ + p)) * D_ + h * 64 + nt * 16 + l15] = acc[r];
      }
    }
  }
}

// ---------------- launcher ----------------
extern "C" void kernel_launch(void* const* d_in, const int* in_sizes, int n_in,
                              void* d_out, int out_size, void* d_ws, size_t ws_size,
                              hipStream_t stream) {
  (void)in_sizes; (void)n_in; (void)out_size; (void)ws_size;
  const float* q  = (const float*)d_in[0];
  const float* k  = (const float*)d_in[1];
  const float* v  = (const float*)d_in[2];
  const float* wq = (const float*)d_in[3];
  const float* wk = (const float*)d_in[4];
  const float* wv = (const float*)d_in[5];
  const int*   ix = (const int*)d_in[6];
  float* out = (float*)d_out;

  // workspace carve-out (256B aligned)
  char* base = (char*)d_ws;
  size_t off = 0;
  auto carve = [&](size_t bytes) -> char* {
    char* p = base + off;
    off = (off + bytes + 255) & ~(size_t)255;
    return p;
  };
  const size_t nBH  = (size_t)B_ * H_;
  const size_t nHd  = nBH * P_ * 64;         // per-head element count
  bf16_t* qh   = (bf16_t*)carve(nHd * 2);
  bf16_t* kh   = (bf16_t*)carve(nHd * 2);
  bf16_t* vh   = (bf16_t*)carve(nHd * 2);
  float*  vhf  = (float*) carve(nHd * 4);
  bf16_t* repb = (bf16_t*)carve((size_t)H_ * T_ * 64 * 2);
  float*  repf = (float*) carve((size_t)H_ * T_ * 64 * 4);
  bf16_t* attn = (bf16_t*)carve(nBH * P_ * PK_ * 2);
  bf16_t* vcT  = (bf16_t*)carve(nBH * 64 * PK_ * 2);

  // 1) RMSNorm + split + bf16
  k_rmsnorm<<<B_ * P_, 256, 0, stream>>>(q, k, v, wq, wk, wv, qh, kh, vh, vhf);
  // 2) rep gather
  {
    int n = H_ * T_ * 64;
    k_repgather<<<(n + 255) / 256, 256, 0, stream>>>(vhf, ix, repb, repf);
  }
  // 2b) zero v_combT padding region (whole buffer, deterministic)
  {
    int n = (int)(nBH * 64 * PK_);
    k_zero16<<<(n + 255) / 256, 256, 0, stream>>>((unsigned short*)vcT, n);
  }
  // 3) attention probabilities
  k_attn<<<(int)(nBH * NT_), 256, 0, stream>>>(qh, kh, attn);
  // 4) sim + top-k sparse softmax + v_aligned + combine
  k_sim<<<(int)(nBH * NT_), 256, 0, stream>>>(vh, repb, repf, vhf, vcT);
  // 5) final GEMM
  k_out<<<(int)nBH, 256, 0, stream>>>(attn, vcT, out);
}
